// HybridTriAxialBlock_1932735283305
// MI455X (gfx1250) — compile-verified, matrix operand order checked by reading, not run on hardware
//
#include <hip/hip_runtime.h>
#include <hip/hip_bf16.h>

// ---------------- problem constants ----------------
#define BB     2048
#define LL     24
#define CELLS  96
#define DD     512
#define KPD    256          // DD/2 packed-k pairs
#define FPITCH 517          // fp32 LDS row pitch
#define ARSTR  260          // packed fragment-major LDS row stride (dwords); 260%64=4 ->
                            // each 16-lane half's b128 load covers all 64 banks once
#define ROWS   48           // 2 batches * 24 line rows
#define EPSLN  1e-5f

typedef __attribute__((ext_vector_type(16))) __bf16 v16bf;
typedef __attribute__((ext_vector_type(8)))  float  v8f;

union FragU { uint4 q[2]; unsigned u[8]; v16bf v; };

// dynamic LDS:
//   R1   : fp32  [ROWS][FPITCH]  (C sums -> H -> Y)
//   PKhi : dword [ROWS][ARSTR]   (fragment-major packed bf16-hi; C then H)
//   PKlo : dword [ROWS][ARSTR]
//   CgS/HgS/YgS : fp32 [2][512] each ; lineOfS[96] int ; invLenS[24] float
#define SMEM_BYTES ((ROWS*FPITCH)*4 + 2*(ROWS*ARSTR)*4 + 3*1024*4 + 96*4 + 24*4)

__device__ __forceinline__ unsigned pack2(float a, float b) {
    // dword = {bf16(a) in [15:0], bf16(b) in [31:16]} (truncation split)
    return (__float_as_uint(a) >> 16) | (__float_as_uint(b) & 0xffff0000u);
}
__device__ __forceinline__ float residf(float a) {
    return a - __uint_as_float(__float_as_uint(a) & 0xffff0000u);
}
// branch-free numerically-stable mish: x * tanh(softplus(x))
__device__ __forceinline__ float mishf(float x) {
    float sp = fmaxf(x, 0.0f) + __logf(1.0f + __expf(-fabsf(x)));   // sp >= 0
    float u  = __expf(-2.0f * sp);                                  // in (0,1]
    return x * (1.0f - u) * __builtin_amdgcn_rcpf(1.0f + u);
}
__device__ __forceinline__ float waveReduceSum(float v) {
    #pragma unroll
    for (int o = 16; o > 0; o >>= 1) v += __shfl_xor(v, o, 32);
    return v;
}

// ---------------- prep: M-derived maps + fragment-major W packing ----------------
// W fragment layout: F[((n*16 + ks)*2 + h)*8 + d] = {bf16(W[2kp][n]), bf16(W[2kp+1][n])}
// with kp = ks*16 + 4h + (d<4 ? d : d+4)  -> one B-fragment = 8 contiguous dwords.
__global__ void tri_prep_kernel(const float* __restrict__ M,
                                const float* __restrict__ W1,
                                const float* __restrict__ W2,
                                int* __restrict__ lineOf, float* __restrict__ invLen,
                                unsigned* __restrict__ W1hi, unsigned* __restrict__ W1lo,
                                unsigned* __restrict__ W2hi, unsigned* __restrict__ W2lo) {
    if (blockIdx.x < 1024) {
        int gid = blockIdx.x * 256 + threadIdx.x;     // [0, 2*131072)
        int mat = gid >> 17;                          // 0 -> W1, 1 -> W2
        int rem = gid & 131071;
        int d   = rem & 7;
        int h   = (rem >> 3) & 1;
        int ks  = (rem >> 4) & 15;
        int n   = rem >> 8;
        int off = (d < 4) ? d : d + 4;
        int kp  = ks * 16 + 4 * h + off;
        const float* S = mat ? W2 : W1;
        unsigned* DH = mat ? W2hi : W1hi;
        unsigned* DL = mat ? W2lo : W1lo;
        float f0 = S[(size_t)(2 * kp) * DD + n];
        float f1 = S[(size_t)(2 * kp + 1) * DD + n];
        DH[rem] = pack2(f0, f1);
        DL[rem] = pack2(residf(f0), residf(f1));
    } else {
        int t = threadIdx.x;
        if (t < CELLS) {                               // lineOf[m]: owning line (partition M)
            int l = 0;
            for (int j = 0; j < LL; ++j)
                if (M[j * CELLS + t] > 0.5f) { l = j; break; }
            lineOf[t] = l;
        }
        if (t >= 128 && t < 128 + LL) {                // 1/line_length
            int l = t - 128;
            float s = 0.f;
            for (int m = 0; m < CELLS; ++m) s += M[l * CELLS + m];
            invLen[l] = (s > 0.f) ? 1.0f / s : 0.f;
        }
    }
}

// pack fp32 LDS buffer -> fragment-major hi/lo bf16 packed LDS (optional per-row scale)
// A fragment layout per row: [ks][h][8 dwords], row stride ARSTR.
__device__ __forceinline__ void pack_buf(const float* __restrict__ src,
                                         unsigned* __restrict__ hi,
                                         unsigned* __restrict__ lo,
                                         const float* __restrict__ scaleRow) {
    for (int i = threadIdx.x; i < ROWS * KPD; i += 256) {
        const int r   = i >> 8, kp = i & 255;
        const int ks  = kp >> 4, r16 = kp & 15;
        const int h   = (r16 >> 2) & 1;                 // {0-3,8-11}->h0, {4-7,12-15}->h1
        const int d   = (r16 & 3) + ((r16 >> 3) << 2);
        const int dst = r * ARSTR + (ks * 2 + h) * 8 + d;
        const float s = scaleRow ? scaleRow[r % LL] : 1.0f;
        float f0 = src[r * FPITCH + 2 * kp] * s;
        float f1 = src[r * FPITCH + 2 * kp + 1] * s;
        hi[dst] = pack2(f0, f1);
        lo[dst] = pack2(residf(f0), residf(f1));
    }
}

// ---------------- bf16x3 WMMA GEMM: [48 x 512] @ [512 x 512] ----------------
// One wave owns a 16-col tile; B fragments (2x b128 hi + 2x b128 lo) loaded once per
// K-step feed 3 row-tile accumulators (9 v_wmma per B-fragment load).
template <bool DOMISH>
__device__ __forceinline__ void gemm_block(const unsigned* __restrict__ Ahi,  // LDS packed
                                           const unsigned* __restrict__ Alo,
                                           const unsigned* __restrict__ Whi,  // global packed
                                           const unsigned* __restrict__ Wlo,
                                           const float* __restrict__ bias,
                                           float* __restrict__ Obuf) {        // fp32 LDS
    const int tid  = threadIdx.x;
    const int wave = tid >> 5;
    const int lane = tid & 31;
    const int lrow = lane & 15;
    const int hsel = lane >> 4;            // fragment half select

    for (int ct = wave; ct < 32; ct += 8) {
        const int n = ct * 16 + lrow;
        v8f acc[3];
        #pragma unroll
        for (int rt = 0; rt < 3; ++rt) acc[rt] = (v8f){0.f,0.f,0.f,0.f,0.f,0.f,0.f,0.f};

        for (int ks = 0; ks < 16; ++ks) {  // K = 512 in steps of 32
            const int fsel  = ks * 2 + hsel;
            const int wbase = (n * 16 + ks) * 16 + hsel * 8;
            FragU bh, bl2;
            bh.q[0]  = *(const uint4*)(Whi + wbase);
            bh.q[1]  = *(const uint4*)(Whi + wbase + 4);
            bl2.q[0] = *(const uint4*)(Wlo + wbase);
            bl2.q[1] = *(const uint4*)(Wlo + wbase + 4);
            #pragma unroll
            for (int rt = 0; rt < 3; ++rt) {
                const unsigned* Arh = Ahi + (rt * 16 + lrow) * ARSTR + fsel * 8;
                const unsigned* Arl = Alo + (rt * 16 + lrow) * ARSTR + fsel * 8;
                FragU ah, al;
                ah.q[0] = *(const uint4*)(Arh);
                ah.q[1] = *(const uint4*)(Arh + 4);
                al.q[0] = *(const uint4*)(Arl);
                al.q[1] = *(const uint4*)(Arl + 4);
                acc[rt] = __builtin_amdgcn_wmma_f32_16x16x32_bf16(false, ah.v, false, bh.v,
                                                                  (short)0, acc[rt], false, false);
                acc[rt] = __builtin_amdgcn_wmma_f32_16x16x32_bf16(false, ah.v, false, bl2.v,
                                                                  (short)0, acc[rt], false, false);
                acc[rt] = __builtin_amdgcn_wmma_f32_16x16x32_bf16(false, al.v, false, bh.v,
                                                                  (short)0, acc[rt], false, false);
            }
        }
        const float bc = bias[n];
        #pragma unroll
        for (int rt = 0; rt < 3; ++rt) {
            const int rbase = rt * 16 + (hsel << 3);
            #pragma unroll
            for (int i = 0; i < 8; ++i) {
                float v = acc[rt][i] + bc;
                if (DOMISH) v = mishf(v);
                Obuf[(rbase + i) * FPITCH + n] = v;
            }
        }
    }
}

// ---------------- fused main kernel: 2 batches per workgroup ----------------
__global__ __launch_bounds__(256)
void tri_main_kernel(const float* __restrict__ x,
                     const float* __restrict__ G1, const float* __restrict__ gb1,
                     const float* __restrict__ G2, const float* __restrict__ gb2,
                     const float* __restrict__ b1, const float* __restrict__ b2,
                     const float* __restrict__ gamma, const float* __restrict__ beta,
                     const unsigned* __restrict__ W1hi, const unsigned* __restrict__ W1lo,
                     const unsigned* __restrict__ W2hi, const unsigned* __restrict__ W2lo,
                     const int* __restrict__ lineOf, const float* __restrict__ invLen,
                     float* __restrict__ out) {
    extern __shared__ char smem_raw[];
    float*    R1   = (float*)smem_raw;                    // [ROWS][FPITCH] fp32: C -> H -> Y
    unsigned* PKhi = (unsigned*)(R1 + ROWS * FPITCH);     // [ROWS][ARSTR]
    unsigned* PKlo = PKhi + ROWS * ARSTR;
    float*    CgS  = (float*)(PKlo + ROWS * ARSTR);       // [2][512]
    float*    HgS  = CgS + 1024;
    float*    YgS  = HgS + 1024;
    int*      lineOfS = (int*)(YgS + 1024);
    float*    invLenS = (float*)(lineOfS + CELLS);

    const int tid = threadIdx.x;
    const int b0  = blockIdx.x * 2;

    if (tid < CELLS) lineOfS[tid] = lineOf[tid];
    if (tid < LL)    invLenS[tid] = invLen[tid];
    for (int i = tid; i < ROWS * FPITCH; i += 256) R1[i] = 0.f;
    __syncthreads();

    // ---- phase 1: line sums + global mean (coalesced over k) ----
    for (int bl = 0; bl < 2; ++bl) {
        const float* xb = x + (size_t)(b0 + bl) * CELLS * DD;
        for (int kb = 0; kb < 2; ++kb) {
            const int k = kb * 256 + tid;
            float tot = 0.f;
            for (int m = 0; m < CELLS; ++m) {
                float v = xb[(size_t)m * DD + k];
                R1[(bl * LL + lineOfS[m]) * FPITCH + k] += v;
                tot += v;
            }
            CgS[bl * DD + k] = tot * (1.0f / (float)CELLS);
        }
    }
    __syncthreads();

    // ---- pack C (folds 1/line_length scaling) + strategic layer 1 ----
    pack_buf(R1, PKhi, PKlo, invLenS);
    for (int i = 0; i < 2; ++i) {
        const int n = tid + 256 * i;
        float a0 = gb1[n], a1 = a0;
        for (int k = 0; k < DD; ++k) {
            const float g = G1[(size_t)k * DD + n];
            a0 += CgS[k] * g;
            a1 += CgS[DD + k] * g;
        }
        HgS[n] = mishf(a0);
        HgS[DD + n] = mishf(a1);
    }
    __syncthreads();

    // ---- strategic layer 2 ----
    for (int i = 0; i < 2; ++i) {
        const int n = tid + 256 * i;
        float a0 = gb2[n], a1 = a0;
        for (int k = 0; k < DD; ++k) {
            const float g = G2[(size_t)k * DD + n];
            a0 += HgS[k] * g;
            a1 += HgS[DD + k] * g;
        }
        YgS[n] = a0;
        YgS[DD + n] = a1;
    }
    __syncthreads();

    // ---- GEMM1: H = mish(C @ W1 + b1) -> R1 ----
    gemm_block<true>(PKhi, PKlo, W1hi, W1lo, b1, R1);
    __syncthreads();

    // ---- pack H ----
    pack_buf(R1, PKhi, PKlo, nullptr);
    __syncthreads();

    // ---- GEMM2: Y = H @ W2 + b2 -> R1 ----
    gemm_block<false>(PKhi, PKlo, W2hi, W2lo, b2, R1);
    __syncthreads();

    // ---- scatter + residual + strategic + LayerNorm (1 wave / row) ----
    const int wave = tid >> 5, lane = tid & 31;
    for (int r = wave; r < 2 * CELLS; r += 8) {
        const int bl = r / CELLS, m = r % CELLS;
        const float* Yrow = R1 + (bl * LL + lineOfS[m]) * FPITCH;
        const float* Ygr  = YgS + bl * DD;
        const size_t rowoff = (size_t)((b0 + bl) * CELLS + m) * DD;
        const float* xr   = x + rowoff;
        float*       orow = out + rowoff;

        float vals[16], s1 = 0.f, s2 = 0.f;
        #pragma unroll
        for (int e = 0; e < 16; ++e) {
            const int d = lane + 32 * e;
            const float v = xr[d] + Yrow[d] + Ygr[d];
            vals[e] = v; s1 += v; s2 += v * v;
        }
        s1 = waveReduceSum(s1);
        s2 = waveReduceSum(s2);
        const float mu  = s1 * (1.0f / (float)DD);
        const float var = s2 * (1.0f / (float)DD) - mu * mu;
        const float rs  = rsqrtf(var + EPSLN);
        #pragma unroll
        for (int e = 0; e < 16; ++e) {
            const int d = lane + 32 * e;
            orow[d] = (vals[e] - mu) * rs * gamma[d] + beta[d];
        }
    }
}

// ---------------- host entry ----------------
extern "C" void kernel_launch(void* const* d_in, const int* in_sizes, int n_in,
                              void* d_out, int out_size, void* d_ws, size_t ws_size,
                              hipStream_t stream) {
    (void)in_sizes; (void)n_in; (void)out_size; (void)ws_size;
    const float* x     = (const float*)d_in[0];
    const float* M     = (const float*)d_in[1];
    const float* W1    = (const float*)d_in[2];
    const float* b1    = (const float*)d_in[3];
    const float* W2    = (const float*)d_in[4];
    const float* b2    = (const float*)d_in[5];
    const float* G1    = (const float*)d_in[6];
    const float* gb1   = (const float*)d_in[7];
    const float* G2    = (const float*)d_in[8];
    const float* gb2   = (const float*)d_in[9];
    const float* gamma = (const float*)d_in[10];
    const float* beta  = (const float*)d_in[11];
    float* out = (float*)d_out;

    char* ws = (char*)d_ws;                       // needs ~2.1 MB
    int*      lineOf = (int*)ws;                  // 96 ints
    float*    invLen = (float*)(ws + 512);        // 24 floats
    unsigned* W1hi   = (unsigned*)(ws + 4096);    // 4 x 512 KB packed weights
    unsigned* W1lo   = W1hi + KPD * DD;
    unsigned* W2hi   = W1lo + KPD * DD;
    unsigned* W2lo   = W2hi + KPD * DD;

    tri_prep_kernel<<<1025, 256, 0, stream>>>(M, W1, W2, lineOf, invLen,
                                              W1hi, W1lo, W2hi, W2lo);
    tri_main_kernel<<<BB / 2, 256, SMEM_BYTES, stream>>>(
        x, G1, gb1, G2, gb2, b1, b2, gamma, beta,
        W1hi, W1lo, W2hi, W2lo, lineOf, invLen, out);
}
